// Rank2Block_15006615734320
// MI455X (gfx1250) — matrix operand order repaired and checked by reading, not compiled, not run on hardware
//
#include <hip/hip_runtime.h>
#include <hip/hip_bf16.h>

typedef __attribute__((ext_vector_type(16))) _Float16 v16h;
typedef __attribute__((ext_vector_type(8)))  float    v8f;

#define WAVES 8
#define TPB   (WAVES * 32)

// ---------------------------------------------------------------------------
// Zero workspace (node/graph accumulators) — run every launch (deterministic).
// ---------------------------------------------------------------------------
__global__ void zero_ws_kernel(float* ws, int n) {
    int i = blockIdx.x * blockDim.x + threadIdx.x;
    int stride = gridDim.x * blockDim.x;
    for (; i < n; i += stride) ws[i] = 0.0f;
}

// ---------------------------------------------------------------------------
// Main kernel: per-16-edge tile,
//   Y = X @ W1 via v_wmma_f32_16x16x32_f16 (8 WMMAs per tile),
//   then 6 unique symmetric components: sum_i silu(outer_c*y_i + b1_i)*W2_i + b2
//   scatter-add into node sums/counts.
// ---------------------------------------------------------------------------
__launch_bounds__(TPB)
__global__ void edge_mlp_scatter(const float* __restrict__ edge_vec,   // [E,3]
                                 const float* __restrict__ x_edge,     // [E,64]
                                 const int*   __restrict__ edge_index, // [E]
                                 const float* __restrict__ W1,         // [64,64] (K=i rows, N=o cols)
                                 const float* __restrict__ b1,         // [64]
                                 const float* __restrict__ W2,         // [64,1]
                                 const float* __restrict__ b2,         // [1]
                                 float* __restrict__ nodeSum,          // [N,6]
                                 float* __restrict__ nodeCnt,          // [N]
                                 int E) {
    __shared__ float sW1[64 * 64];
    __shared__ float sB1[64];
    __shared__ float sW2[64];
    __shared__ float sY[WAVES][16][68];   // per-wave Y tile, padded stride 68

    const int tid  = threadIdx.x;
    const int lane = tid & 31;
    const int wave = tid >> 5;
    const int mrow = lane & 15;   // row / N-within-tile index for this lane
    const int hi   = lane >> 4;   // half-wave selector

    // Stage weights into LDS (W1 is L2-hot across blocks; this is cheap).
    for (int i = tid; i < 64 * 64; i += TPB) sW1[i] = W1[i];
    if (tid < 64) { sB1[tid] = b1[tid]; sW2[tid] = W2[tid]; }
    __syncthreads();

    // -----------------------------------------------------------------
    // Pack W1 into WMMA B-fragments (f16). B is K x N; for 16x16x32 f16:
// lanes 0-15 hold K=0..7,16..23; lanes 16-31 hold K=8..15,24..31 (matches A).
    // 4 N-blocks x 2 K-blocks covering the full 64x64.
    // -----------------------------------------------------------------
    v16h Bf[4][2];
    #pragma unroll
    for (int nb = 0; nb < 4; ++nb) {
        #pragma unroll
        for (int kb = 0; kb < 2; ++kb) {
            const int n     = nb * 16 + mrow;
            const int kbase = kb * 32 + hi * 8;
            v16h b;
            #pragma unroll
            for (int j = 0; j < 16; ++j)
                b[j] = (_Float16)sW1[(kbase + (j & 7) + ((j >> 3) << 4)) * 64 + n];
            Bf[nb][kb] = b;
        }
    }

    const float b2v    = b2[0];
    const int   tiles  = (E + 15) >> 4;
    const int   wgid   = blockIdx.x * WAVES + wave;
    const int   wstep  = gridDim.x * WAVES;

    for (int t = wgid; t < tiles; t += wstep) {
        const int ebase = t << 4;

        // ---------------- A fragments: 16 rows x 64 K (two 16x32 f16 frags).
        // 16-bit A layout: lanes 0-15 = rows, VGPR0-3 -> K 0..7, VGPR4-7 -> K 16..23
        // (lanes 16-31 shifted by +8 within each 16-K group).
        int arow = ebase + mrow; if (arow >= E) arow = E - 1;
        const float* xr = x_edge + (size_t)arow * 64;
        v16h A[2];
        #pragma unroll
        for (int kb = 0; kb < 2; ++kb) {
            v16h a;
            #pragma unroll
            for (int bj = 0; bj < 2; ++bj) {
                const int ks = kb * 32 + bj * 16 + hi * 8;
                const float4 f0 = *(const float4*)(xr + ks);
                const float4 f1 = *(const float4*)(xr + ks + 4);
                a[bj * 8 + 0] = (_Float16)f0.x; a[bj * 8 + 1] = (_Float16)f0.y;
                a[bj * 8 + 2] = (_Float16)f0.z; a[bj * 8 + 3] = (_Float16)f0.w;
                a[bj * 8 + 4] = (_Float16)f1.x; a[bj * 8 + 5] = (_Float16)f1.y;
                a[bj * 8 + 6] = (_Float16)f1.z; a[bj * 8 + 7] = (_Float16)f1.w;
            }
            A[kb] = a;
        }

        // ---------------- Y = X @ W1  (16 x 64), spill to per-wave LDS.
        // C/D layout: VGPR r -> row r (lanes 0-15) / row r+8 (lanes 16-31), col = lane%16.
        #pragma unroll
        for (int nb = 0; nb < 4; ++nb) {
            v8f c = {};
            c = __builtin_amdgcn_wmma_f32_16x16x32_f16(false, A[0], false, Bf[nb][0],
                                                       (short)0, c, false, false);
            c = __builtin_amdgcn_wmma_f32_16x16x32_f16(false, A[1], false, Bf[nb][1],
                                                       (short)0, c, false, false);
            #pragma unroll
            for (int r = 0; r < 8; ++r)
                sY[wave][r + hi * 8][nb * 16 + mrow] = c[r];
        }
        // In-wave cross-lane visibility: LDS is in-order per wave; just drain DScnt.
        asm volatile("s_wait_dscnt 0x0" ::: "memory");

        // ---------------- Per-edge epilogue: 6 unique symmetric components.
        const int  e  = ebase + mrow;
        const bool ok = (e < E);
        int   nidx = 0;
        float o[6] = {0.f, 0.f, 0.f, 0.f, 0.f, 0.f};
        if (ok) {
            nidx = edge_index[e];
            const float d0 = edge_vec[(size_t)e * 3 + 0];
            const float d1 = edge_vec[(size_t)e * 3 + 1];
            const float d2 = edge_vec[(size_t)e * 3 + 2];
            o[0] = d0 * d0; o[1] = d0 * d1; o[2] = d0 * d2;
            o[3] = d1 * d1; o[4] = d1 * d2; o[5] = d2 * d2;
        }

        const float* yrow = &sY[wave][mrow][0];
        #pragma unroll
        for (int p = 0; p < 3; ++p) {
            const int   c  = p * 2 + hi;     // lanes 0-15: even comp, 16-31: odd comp
            const float oc = o[c];
            float acc = 0.0f;
            #pragma unroll 8
            for (int i = 0; i < 64; ++i) {
                const float tt  = fmaf(oc, yrow[i], sB1[i]);
                const float sig = __builtin_amdgcn_rcpf(1.0f + __expf(-tt));
                acc = fmaf(tt * sig, sW2[i], acc);
            }
            if (ok) {
                atomicAdd(&nodeSum[(size_t)nidx * 6 + c], acc + b2v);
                if (p == 0 && hi == 0) atomicAdd(&nodeCnt[nidx], 1.0f);
            }
        }
    }
}

// ---------------------------------------------------------------------------
// Node -> graph segment mean; per-block LDS accumulation (G <= 16), one global
// atomic per slot per block.
// ---------------------------------------------------------------------------
__global__ void node_to_graph(const float* __restrict__ nodeSum,
                              const float* __restrict__ nodeCnt,
                              const int*   __restrict__ batch,
                              int N,
                              float* __restrict__ graphSum,   // [G,6]
                              float* __restrict__ graphCnt) { // [G]
    __shared__ float gs[16 * 6];
    __shared__ float gc[16];
    const int tid = threadIdx.x;
    if (tid < 16 * 6) gs[tid] = 0.0f;
    if (tid < 16)     gc[tid] = 0.0f;
    __syncthreads();

    for (int n = blockIdx.x * blockDim.x + tid; n < N; n += gridDim.x * blockDim.x) {
        const int   g   = batch[n] & 15;
        const float cnt = nodeCnt[n];
        const float inv = 1.0f / fmaxf(cnt, 1.0f);
        #pragma unroll
        for (int c = 0; c < 6; ++c)
            atomicAdd(&gs[g * 6 + c], nodeSum[(size_t)n * 6 + c] * inv);
        atomicAdd(&gc[g], 1.0f);
    }
    __syncthreads();
    if (tid < 16 * 6) atomicAdd(&graphSum[tid], gs[tid]);
    if (tid < 16)     atomicAdd(&graphCnt[tid], gc[tid]);
}

// ---------------------------------------------------------------------------
// Finalize: divide by node count and expand 6 symmetric comps -> 9 output.
// ---------------------------------------------------------------------------
__global__ void finalize_out(const float* __restrict__ graphSum,
                             const float* __restrict__ graphCnt,
                             float* __restrict__ out, int G) {
    const int g = blockIdx.x * blockDim.x + threadIdx.x;
    if (g < G) {
        const float inv = 1.0f / fmaxf(graphCnt[g], 1.0f);
        const float o0 = graphSum[g * 6 + 0] * inv;
        const float o1 = graphSum[g * 6 + 1] * inv;
        const float o2 = graphSum[g * 6 + 2] * inv;
        const float o3 = graphSum[g * 6 + 3] * inv;
        const float o4 = graphSum[g * 6 + 4] * inv;
        const float o5 = graphSum[g * 6 + 5] * inv;
        float* og = out + g * 9;
        og[0] = o0; og[1] = o1; og[2] = o2;
        og[3] = o1; og[4] = o3; og[5] = o4;
        og[6] = o2; og[7] = o4; og[8] = o5;
    }
}

// ---------------------------------------------------------------------------
extern "C" void kernel_launch(void* const* d_in, const int* in_sizes, int n_in,
                              void* d_out, int out_size, void* d_ws, size_t ws_size,
                              hipStream_t stream) {
    const float* edge_vec   = (const float*)d_in[0];
    const float* x_edge     = (const float*)d_in[1];
    const int*   edge_index = (const int*)  d_in[2];
    const int*   batch      = (const int*)  d_in[3];
    // d_in[4] = n_nodes, d_in[5] = n_graphs (device scalars; sizes used instead)
    const float* W1 = (const float*)d_in[6];
    const float* b1 = (const float*)d_in[7];
    const float* W2 = (const float*)d_in[8];
    const float* b2 = (const float*)d_in[9];

    const int E = in_sizes[2];        // edge_index has E entries
    const int N = in_sizes[3];        // batch has N entries
    const int G = out_size / 9;       // output is [G, 9]

    float* ws        = (float*)d_ws;
    float* nodeSum   = ws;                         // N*6
    float* nodeCnt   = nodeSum + (size_t)N * 6;    // N
    float* graphSum  = nodeCnt + N;                // G*6
    float* graphCnt  = graphSum + (size_t)G * 6;   // G
    const int wsFloats = N * 7 + G * 7;

    zero_ws_kernel<<<256, 256, 0, stream>>>(ws, wsFloats);

    const int tiles = (E + 15) / 16;
    int blocks = (tiles + WAVES - 1) / WAVES;
    if (blocks > 1024) blocks = 1024;
    if (blocks < 1) blocks = 1;
    edge_mlp_scatter<<<blocks, TPB, 0, stream>>>(edge_vec, x_edge, edge_index,
                                                 W1, b1, W2, b2,
                                                 nodeSum, nodeCnt, E);

    int nblocks = (N + 255) / 256;
    if (nblocks > 512) nblocks = 512;
    if (nblocks < 1) nblocks = 1;
    node_to_graph<<<nblocks, 256, 0, stream>>>(nodeSum, nodeCnt, batch, N,
                                               graphSum, graphCnt);

    finalize_out<<<1, 64, 0, stream>>>(graphSum, graphCnt, (float*)d_out, G);
}
